// DynamicNeuralTextureExpression_88957362634865
// MI455X (gfx1250) — matrix operand order repaired and checked by reading, not compiled, not run on hardware
//
#include <hip/hip_runtime.h>
#include <hip/hip_bf16.h>
#include <hip/hip_fp16.h>

typedef __attribute__((ext_vector_type(16))) _Float16 v16h;
typedef __attribute__((ext_vector_type(8)))  float    v8f;
typedef __attribute__((ext_vector_type(4)))  float    v4f;
typedef __attribute__((ext_vector_type(2)))  __fp16   fp16x2;
typedef __attribute__((ext_vector_type(4)))  unsigned u4v;

#define IN_SIZE 76
#define NPIX    1048576       // 1024*1024
#define TSW     1088          // padded tex width  (17*64)
#define TSH     1040          // padded tex height (65*16)
#define CROWS   20            // input tile rows  (16 out rows + 3 halo, rounded)
#define CCOLS   68            // input tile cols  (64 out cols + 3 halo, rounded)

// ---------------- Kernel 1: MLP layers 1+2 (trivial, one block) ----------------
__global__ void k_mlp12(const float* __restrict__ x,
                        const float* __restrict__ W1, const float* __restrict__ b1,
                        const float* __restrict__ W2, const float* __restrict__ b2,
                        float* __restrict__ h2out) {
  __shared__ float xs[IN_SIZE];
  __shared__ float h1[128];
  int t = threadIdx.x;
  if (t < IN_SIZE) xs[t] = x[t];
  __syncthreads();
  {
    const float* w = W1 + t * IN_SIZE;
    float acc = b1[t];
    #pragma unroll 4
    for (int i = 0; i < IN_SIZE; ++i) acc += w[i] * xs[i];
    h1[t] = acc >= 0.0f ? acc : 0.02f * acc;
  }
  __syncthreads();
  if (t < 64) {
    const float* w = W2 + t * 128;
    float acc = b2[t];
    #pragma unroll 4
    for (int i = 0; i < 128; ++i) acc += w[i] * h1[i];
    h2out[t] = acc >= 0.0f ? acc : 0.02f * acc;
  }
}

// -------- Kernel 2: MLP layer 3 + tanh -> filter in WMMA A-fragment order --------
// A-fragment (16x16x32 f16, contiguous K order, same as B): lane = g*16 + oc (g=lane>>4, M=oc=lane&15),
// element i <-> K: K = i + g*16.  Step s = kh*4+kw, K = ic.
__global__ void k_filter(const float* __restrict__ h2,
                         const float* __restrict__ W3, const float* __restrict__ b3,
                         _Float16* __restrict__ afrag) {
  __shared__ float hs[64];
  int t = threadIdx.x;
  if (t < 64) hs[t] = h2[t];
  __syncthreads();
  int o = blockIdx.x * 256 + t;              // 0..8191 = ((oc*32+ic)*4+kh)*4+kw
  const float* w = W3 + o * 64;
  float acc = b3[o];
  #pragma unroll 4
  for (int i = 0; i < 64; ++i) acc += w[i] * hs[i];
  float f = tanhf(acc);
  int kw = o & 3, kh = (o >> 2) & 3, ic = (o >> 4) & 31, oc = o >> 9;
  int s = kh * 4 + kw;
  int g = ic >> 4;
  int i = ic & 15;
  int lane = g * 16 + oc;
  afrag[(s * 32 + lane) * 16 + i] = (_Float16)f;
}

// ---------------- Kernel 3: implicit-GEMM conv via WMMA f16 ----------------
// Grid (17,65), 512 threads (16 waves). Wave w: output row oh0+w, cols ow0..ow0+63
// as 4 tiles of 16 px; 16 (kh,kw) steps, K=32 channels per step.
__global__ __launch_bounds__(512) void k_conv(const float* __restrict__ data,
                                              const _Float16* __restrict__ afrag,
                                              float* __restrict__ tex) {
  __shared__ __align__(16) unsigned char smem[CROWS * CCOLS * 32 * 2]; // 87040 B
  _Float16* sD = (_Float16*)smem;                 // [row][col][ch] f16
  const int tid  = threadIdx.x;
  const int wave = tid >> 5;
  const int lane = tid & 31;
  const int g    = lane >> 4;
  const int n    = lane & 15;
  const int ow0  = blockIdx.x * 64;
  const int oh0  = blockIdx.y * 16;

  // Loader: each thread owns one (row,col) pixel, packs all 32 channels.
  // Global reads coalesced across threads (consecutive cols); LDS writes are
  // four contiguous ds_store_b128 (64B/thread).
  for (int pos = tid; pos < CROWS * CCOLS; pos += 512) {
    const int row = pos / CCOLS;
    const int col = pos - row * CCOLS;
    const int y = oh0 - 2 + row;
    const int x = ow0 - 2 + col;
    const bool inb = ((unsigned)y < 1024u) && ((unsigned)x < 1024u);
    const int base = y * 1024 + x;
    unsigned pk[16];
    #pragma unroll
    for (int c2 = 0; c2 < 16; ++c2) {
      float v0 = 0.0f, v1 = 0.0f;
      if (inb) {
        v0 = data[(2 * c2) * NPIX + base];
        v1 = data[(2 * c2 + 1) * NPIX + base];
      }
      fp16x2 p = __builtin_amdgcn_cvt_pkrtz(v0, v1);  // v_cvt_pk_rtz_f16_f32
      pk[c2] = __builtin_bit_cast(unsigned, p);
    }
    unsigned* dst = (unsigned*)(smem + pos * 64);
    #pragma unroll
    for (int q = 0; q < 4; ++q)
      *(u4v*)(dst + q * 4) = u4v{pk[q*4], pk[q*4+1], pk[q*4+2], pk[q*4+3]};
  }
  __syncthreads();

  v8f acc[4] = {};
  for (int s = 0; s < 16; ++s) {
    const int kh = s >> 2, kw = s & 3;
    const v16h a = *(const v16h*)(afrag + (s * 32 + lane) * 16); // hot in L2
    const int rbase = (wave + kh) * CCOLS;
    #pragma unroll
    for (int t4 = 0; t4 < 4; ++t4) {
      const int col = t4 * 16 + n + kw;
      const v16h b = *(const v16h*)(&sD[(rbase + col) * 32 + g * 16]); // 32B/lane
      acc[t4] = __builtin_amdgcn_wmma_f32_16x16x32_f16(
          false, a, false, b, (short)0, acc[t4], false, false);
    }
  }
  __syncthreads();   // all waves done reading sD -> reuse it for store staging

  // C layout: VGPR r of lane -> (oc = r + g*8, pixel = n).  Transpose via LDS,
  // store pixel-interleaved fp32 tex[(ty*TSW+tx)*16 + c] fully coalesced.
  float* my = (float*)smem + wave * 256;          // 16 px * 16 ch per wave
  const int ty = oh0 + wave;
  const int p  = lane >> 1, hf = lane & 1;
  for (int t4 = 0; t4 < 4; ++t4) {
    #pragma unroll
    for (int r = 0; r < 8; ++r)
      my[n * 16 + (r + g * 8)] = acc[t4][r];
    // same-wave LDS RAW: DS ops are in-order per wave; compiler inserts waits
    v4f o0 = *(const v4f*)(my + p * 16 + hf * 8);
    v4f o1 = *(const v4f*)(my + p * 16 + hf * 8 + 4);
    const int tx = ow0 + t4 * 16 + p;
    float* dst = tex + (unsigned)((ty * TSW + tx) * 16 + hf * 8);
    *(v4f*)dst       = o0;
    *((v4f*)dst + 1) = o1;
  }
}

// ---------------- Kernel 4: bilinear grid-sample (border pad) ----------------
__global__ __launch_bounds__(256) void k_sample(const float* __restrict__ uv,
                                                const float* __restrict__ tex,
                                                float* __restrict__ out) {
  const int pix = blockIdx.x * 256 + threadIdx.x;
  const float gx = uv[pix];
  const float gy = uv[NPIX + pix];
  float ix = ((gx + 1.0f) * 1025.0f - 1.0f) * 0.5f;
  float iy = ((gy + 1.0f) * 1025.0f - 1.0f) * 0.5f;
  ix = fminf(fmaxf(ix, 0.0f), 1024.0f);
  iy = fminf(fmaxf(iy, 0.0f), 1024.0f);
  const float fx0 = floorf(ix), fy0 = floorf(iy);
  const float wx = ix - fx0, wy = iy - fy0;
  const int x0 = (int)fx0, y0 = (int)fy0;
  const int x1 = min(x0 + 1, 1024), y1 = min(y0 + 1, 1024);
  const float* p00 = tex + (unsigned)((y0 * TSW + x0) * 16);
  const float* p01 = tex + (unsigned)((y0 * TSW + x1) * 16);
  const float* p10 = tex + (unsigned)((y1 * TSW + x0) * 16);
  const float* p11 = tex + (unsigned)((y1 * TSW + x1) * 16);
  const float w00 = (1.0f - wx) * (1.0f - wy), w01 = wx * (1.0f - wy);
  const float w10 = (1.0f - wx) * wy,          w11 = wx * wy;
  #pragma unroll
  for (int c4 = 0; c4 < 4; ++c4) {
    v4f v00 = *(const v4f*)(p00 + c4 * 4);
    v4f v01 = *(const v4f*)(p01 + c4 * 4);
    v4f v10 = *(const v4f*)(p10 + c4 * 4);
    v4f v11 = *(const v4f*)(p11 + c4 * 4);
    v4f r = v00 * w00 + v01 * w01 + v10 * w10 + v11 * w11;
    #pragma unroll
    for (int j = 0; j < 4; ++j)
      out[(unsigned)((c4 * 4 + j) * NPIX + pix)] = r[j];
  }
}

extern "C" void kernel_launch(void* const* d_in, const int* in_sizes, int n_in,
                              void* d_out, int out_size, void* d_ws, size_t ws_size,
                              hipStream_t stream) {
  const float* expressions = (const float*)d_in[0];
  // d_in[1] audio_features: unused by the reference
  const float* uv   = (const float*)d_in[2];
  const float* data = (const float*)d_in[3];
  const float* W1 = (const float*)d_in[4];
  const float* b1 = (const float*)d_in[5];
  const float* W2 = (const float*)d_in[6];
  const float* b2 = (const float*)d_in[7];
  const float* W3 = (const float*)d_in[8];
  const float* b3 = (const float*)d_in[9];
  float* out = (float*)d_out;

  char* ws = (char*)d_ws;
  float*     h2    = (float*)ws;                  // 64 floats
  _Float16*  afrag = (_Float16*)(ws + 1024);      // 16KB  A-fragments
  float*     tex   = (float*)(ws + 32768);        // TSH*TSW*16 fp32 (~72.4 MB)

  k_mlp12 <<<1, 128, 0, stream>>>(expressions, W1, b1, W2, b2, h2);
  k_filter<<<32, 256, 0, stream>>>(h2, W3, b3, afrag);
  k_conv  <<<dim3(17, 65), 512, 0, stream>>>(data, afrag, tex);
  k_sample<<<4096, 256, 0, stream>>>(uv, tex, out);
}
